// FusedNetwork_65833258713323
// MI455X (gfx1250) — compile-verified, matrix operand order checked by reading, not run on hardware
//
#include <hip/hip_runtime.h>

// Fused coordinate MLP for MI455X (gfx1250, wave32, WMMA).
//   enc(x)[16x64] -> relu(@W0^T+b0)[16x64] -> relu(@W1^T+b1) -> relu(@W2^T+b2) -> @W3^T+b3 [16x4]
// One wave per 16-point tile; all GEMMs via v_wmma_f32_16x16x32_f16.
//
// Cross-lane activation hand-off uses per-wave LDS staging. Ordering is enforced
// with wavefront-scope fences + wave_barrier at phase boundaries: this keeps the
// accesses on the DS pipe (plain ds_store_b16/ds_load_b32, same-wave DS is in-order
// wave-wide per ISA) while stopping LICM from hoisting the loads (R1 bug), and
// avoids the volatile->flat/SCOPE_SYS serialization observed in R2.

#define WAVES 8        // waves per block (256 threads)
#define WS 70          // LDS weight row stride in halves (35-dword rows -> conflict-free)
#define HS 70          // LDS activation-staging row stride in halves

typedef __attribute__((ext_vector_type(16))) _Float16 v16h;
typedef __attribute__((ext_vector_type(8)))  float    v8f;

union Frag { v16h h; unsigned u[8]; };

// Compiler-level phase boundary for same-wave cross-lane LDS communication.
__device__ __forceinline__ void lds_phase_fence() {
  __builtin_amdgcn_fence(__ATOMIC_SEQ_CST, "wavefront");
  __builtin_amdgcn_wave_barrier();
  __builtin_amdgcn_fence(__ATOMIC_SEQ_CST, "wavefront");
}

// WMMA f16 fragment layout (CDNA5 ISA): element e holds K = e + 8*[e>=8] + 8*half.
// Element pairs are K-contiguous -> dword loads.
__device__ __forceinline__ v16h ld_frag(const _Float16* rowp, int kbase, int half8) {
  Frag r;
#pragma unroll
  for (int j = 0; j < 8; ++j) {
    const int e = 2 * j;
    const int k = kbase + e + ((e >= 8) ? 8 : 0) + half8;
    r.u[j] = *(const unsigned*)(rowp + k);          // 4B-aligned: row stride 140B, k even
  }
  return r.h;
}

// NeRF-style encoding channel k of point (x0,x1,x2); channels 51..63 are zero pad.
__device__ __forceinline__ float enc_val(int k, float x0, float x1, float x2) {
  if (k < 3)  return (k == 0) ? x0 : ((k == 1) ? x1 : x2);
  if (k >= 51) return 0.0f;
  int i = k - 3;
  const bool is_cos = (i >= 24);
  if (is_cos) i -= 24;
  const float xv  = (i < 8) ? x0 : ((i < 16) ? x1 : x2);
  const float arg = ldexpf(xv * 3.14159265358979323846f, i & 7);  // x * pi * 2^f
  return is_cos ? __cosf(arg) : __sinf(arg);
}

__global__ __launch_bounds__(256) void coord_mlp_wmma(
    const float* __restrict__ inp,
    const float* __restrict__ W0, const float* __restrict__ b0,
    const float* __restrict__ W1, const float* __restrict__ b1,
    const float* __restrict__ W2, const float* __restrict__ b2,
    const float* __restrict__ W3, const float* __restrict__ b3,
    float* __restrict__ out, int npts)
{
  __shared__ __attribute__((aligned(16))) _Float16 sW0[64 * WS];
  __shared__ __attribute__((aligned(16))) _Float16 sW1[64 * WS];
  __shared__ __attribute__((aligned(16))) _Float16 sW2[64 * WS];
  __shared__ __attribute__((aligned(16))) _Float16 sW3[16 * WS];
  __shared__ __attribute__((aligned(16))) _Float16 sH[WAVES][16 * HS];

  const int tid = threadIdx.x;

  // Stage weights fp32 -> fp16 into LDS once per block (zero-padded).
  for (int idx = tid; idx < 64 * 64; idx += 256) {
    const int o = idx >> 6, k = idx & 63;
    sW0[o * WS + k] = (_Float16)((k < 51) ? W0[o * 51 + k] : 0.0f);
    sW1[o * WS + k] = (_Float16)W1[idx];
    sW2[o * WS + k] = (_Float16)W2[idx];
  }
  for (int idx = tid; idx < 16 * 64; idx += 256) {
    const int o = idx >> 6, k = idx & 63;
    sW3[o * WS + k] = (_Float16)((o < 4) ? W3[o * 64 + k] : 0.0f);
  }
  __syncthreads();

  const int wave  = tid >> 5;
  const int lane  = tid & 31;
  const int l15   = lane & 15;
  const int half8 = (lane >> 4) * 8;
  _Float16* hst = &sH[wave][0];

  // Per-lane bias registers (column = 16*g + l15 in each group).
  float bias0[4], bias1[4], bias2[4];
#pragma unroll
  for (int g = 0; g < 4; ++g) {
    bias0[g] = b0[g * 16 + l15];
    bias1[g] = b1[g * 16 + l15];
    bias2[g] = b2[g * 16 + l15];
  }
  const float bias3 = b3[l15 & 3];

  const int ntiles = npts >> 4;
  const int tstep  = gridDim.x * WAVES;

  for (int t = blockIdx.x * WAVES + wave; t < ntiles; t += tstep) {
    const int base = t << 4;
    if (t + tstep < ntiles)
      __builtin_prefetch(inp + (size_t)(t + tstep) * 48, 0, 3);  // global_prefetch_b8

    const float* ip = inp + (size_t)(base + l15) * 3;
    const float x0 = ip[0];
    const float x1 = ip[1];
    const float x2 = ip[2];

    // Layer-0 A fragments built directly in WMMA A layout (no memory round-trip).
    Frag a0, a1;
#pragma unroll
    for (int e = 0; e < 16; ++e) {
      const int kk = e + ((e >= 8) ? 8 : 0) + half8;
      a0.h[e] = (_Float16)enc_val(kk,      x0, x1, x2);
      a1.h[e] = (_Float16)enc_val(kk + 32, x0, x1, x2);
    }

    // ---- layer 0: H = relu(enc @ W0^T + b0) -> LDS staging ----
#pragma unroll
    for (int g = 0; g < 4; ++g) {
      v8f acc;
#pragma unroll
      for (int i = 0; i < 8; ++i) acc[i] = bias0[g];
      const v16h bLo = ld_frag(&sW0[(g * 16 + l15) * WS], 0,  half8);
      const v16h bHi = ld_frag(&sW0[(g * 16 + l15) * WS], 32, half8);
      acc = __builtin_amdgcn_wmma_f32_16x16x32_f16(false, a0.h, false, bLo, (short)0, acc, false, false);
      acc = __builtin_amdgcn_wmma_f32_16x16x32_f16(false, a1.h, false, bHi, (short)0, acc, false, false);
#pragma unroll
      for (int i = 0; i < 8; ++i)
        hst[(i + half8) * HS + g * 16 + l15] = (_Float16)fmaxf(acc[i], 0.0f);
    }
    lds_phase_fence();   // stores visible before cross-lane reads

    // ---- hidden layers 1 & 2: H = relu(H @ W^T + b), staged through LDS ----
#pragma unroll
    for (int layer = 0; layer < 2; ++layer) {
      const _Float16* sW   = layer ? sW2   : sW1;
      const float*    bias = layer ? bias2 : bias1;
      const v16h hA0 = ld_frag(&hst[l15 * HS], 0,  half8);
      const v16h hA1 = ld_frag(&hst[l15 * HS], 32, half8);
      lds_phase_fence(); // reads done before buffer is overwritten
#pragma unroll
      for (int g = 0; g < 4; ++g) {
        v8f acc;
#pragma unroll
        for (int i = 0; i < 8; ++i) acc[i] = bias[g];
        const v16h bLo = ld_frag(&sW[(g * 16 + l15) * WS], 0,  half8);
        const v16h bHi = ld_frag(&sW[(g * 16 + l15) * WS], 32, half8);
        acc = __builtin_amdgcn_wmma_f32_16x16x32_f16(false, hA0, false, bLo, (short)0, acc, false, false);
        acc = __builtin_amdgcn_wmma_f32_16x16x32_f16(false, hA1, false, bHi, (short)0, acc, false, false);
#pragma unroll
        for (int i = 0; i < 8; ++i)
          hst[(i + half8) * HS + g * 16 + l15] = (_Float16)fmaxf(acc[i], 0.0f);
      }
      lds_phase_fence();
    }

    // ---- output layer: [16x4] = H @ W3^T + b3 (N padded to 16, cols 4..15 dropped) ----
    {
      const v16h oA0 = ld_frag(&hst[l15 * HS], 0,  half8);
      const v16h oA1 = ld_frag(&hst[l15 * HS], 32, half8);
      lds_phase_fence(); // reads done before next tile overwrites staging
      v8f acc;
#pragma unroll
      for (int i = 0; i < 8; ++i) acc[i] = bias3;
      const v16h bLo = ld_frag(&sW3[l15 * WS], 0,  half8);
      const v16h bHi = ld_frag(&sW3[l15 * WS], 32, half8);
      acc = __builtin_amdgcn_wmma_f32_16x16x32_f16(false, oA0, false, bLo, (short)0, acc, false, false);
      acc = __builtin_amdgcn_wmma_f32_16x16x32_f16(false, oA1, false, bHi, (short)0, acc, false, false);
      if (l15 < 4) {
        float* op = out + ((size_t)(base + half8) * 4 + l15);
#pragma unroll
        for (int i = 0; i < 8; ++i)
          op[4 * i] = acc[i];                   // immediate-offset global_store_b32
      }
    }
  }
}

extern "C" void kernel_launch(void* const* d_in, const int* in_sizes, int n_in,
                              void* d_out, int out_size, void* d_ws, size_t ws_size,
                              hipStream_t stream) {
  (void)n_in; (void)out_size; (void)d_ws; (void)ws_size;
  const float* inp = (const float*)d_in[0];
  const float* W0  = (const float*)d_in[1];
  const float* b0  = (const float*)d_in[2];
  const float* W1  = (const float*)d_in[3];
  const float* b1  = (const float*)d_in[4];
  const float* W2  = (const float*)d_in[5];
  const float* b2  = (const float*)d_in[6];
  const float* W3  = (const float*)d_in[7];
  const float* b3  = (const float*)d_in[8];
  float* out = (float*)d_out;

  const int npts = in_sizes[0] / 3;           // [N,3] coords
  const int grid = 1024;                      // 8192 wave streams over 65536 tiles
  hipLaunchKernelGGL(coord_mlp_wmma, dim3(grid), dim3(256), 0, stream,
                     inp, W0, b0, W1, b1, W2, b2, W3, b3, out, npts);
}
